// MultiHeadAttention_26268019982925
// MI455X (gfx1250) — matrix operand order repaired and checked by reading, not compiled, not run on hardware
//
#include <hip/hip_runtime.h>
#include <hip/hip_bf16.h>

typedef __bf16 bf16_t;
typedef __attribute__((ext_vector_type(16))) __bf16 v16bf;
typedef __attribute__((ext_vector_type(8)))  __bf16 v8bf;
typedef __attribute__((ext_vector_type(8)))  float  v8f;
typedef int v4i_t __attribute__((vector_size(16)));

// ---- gfx1250 async HBM->LDS copy (ASYNCcnt-tracked), with sync fallback ----
#if __has_builtin(__builtin_amdgcn_global_load_async_to_lds_b128)
#define HAVE_ASYNC_LDS 1
__device__ __forceinline__ void async_copy16(const bf16_t* g, bf16_t* l) {
  __builtin_amdgcn_global_load_async_to_lds_b128(
      (__attribute__((address_space(1))) v4i_t*)(void*)g,
      (__attribute__((address_space(3))) v4i_t*)(void*)l, 0, 0);
}
#if __has_builtin(__builtin_amdgcn_s_wait_asynccnt)
__device__ __forceinline__ void wait_async0() { __builtin_amdgcn_s_wait_asynccnt(0); }
#else
__device__ __forceinline__ void wait_async0() {
  asm volatile("s_wait_asynccnt 0x0" ::: "memory");
}
#endif
#else
#define HAVE_ASYNC_LDS 0
__device__ __forceinline__ void async_copy16(const bf16_t* g, bf16_t* l) {
  *(v8bf*)l = *(const v8bf*)g;
}
__device__ __forceinline__ void wait_async0() {}
#endif

// D = A(16x32 bf16) * B(32x16 bf16) + C(16x16 f32)
__device__ __forceinline__ v8f wmma_bf16(v16bf a, v16bf b, v8f c) {
  return __builtin_amdgcn_wmma_f32_16x16x32_bf16(false, a, false, b, (short)0, c,
                                                 false, false);
}

// A fragment 16x32 (MxK), source layout [row][k], k contiguous.
// lanes 0-15: M=lane, K = k0+{0..7} and k0+{16..23}; lanes 16-31: +8.
__device__ __forceinline__ v16bf frag_a(const bf16_t* base, int row0, int stride, int k0) {
  const int lane = threadIdx.x & 31;
  const int l = lane & 15, hb = lane >> 4;
  const bf16_t* p = base + (row0 + l) * stride + k0 + (hb << 3);
  v8bf x0 = *(const v8bf*)(p);
  v8bf x1 = *(const v8bf*)(p + 16);
  v16bf a;
#pragma unroll
  for (int i = 0; i < 8; ++i) { a[i] = x0[i]; a[i + 8] = x1[i]; }
  return a;
}

// B fragment 32x16 (KxN), source layout [col][k], k contiguous.
// lanes 0-15: N=lane, K = k0+{0..7} and k0+{16..23}; lanes 16-31: +8.
__device__ __forceinline__ v16bf frag_b(const bf16_t* base, int col0, int stride, int k0) {
  const int lane = threadIdx.x & 31;
  const int nn = lane & 15, hb = lane >> 4;
  const bf16_t* p = base + (col0 + nn) * stride + k0 + (hb << 3);
  v8bf x0 = *(const v8bf*)(p);
  v8bf x1 = *(const v8bf*)(p + 16);
  v16bf b;
#pragma unroll
  for (int i = 0; i < 8; ++i) { b[i] = x0[i]; b[i + 8] = x1[i]; }
  return b;
}

// ---------------------------------------------------------------------------
// Kernel 1: q/k/v = W(64x64) @ X(64x256) + b for one (n, h) row of pixels.
// Output stored bf16 in (N, W, H, C) layout; q pre-scaled by 1/sqrt(W)=1/16.
// ---------------------------------------------------------------------------
__global__ __launch_bounds__(256) void proj_kernel(
    const float* __restrict__ hin,
    const float* __restrict__ wq, const float* __restrict__ bq,
    const float* __restrict__ wk, const float* __restrict__ bk,
    const float* __restrict__ wv, const float* __restrict__ bv,
    bf16_t* __restrict__ qR, bf16_t* __restrict__ kR, bf16_t* __restrict__ vR) {
  __shared__ __align__(16) bf16_t sW[3 * 64 * 72];  // weights, row stride 72
  __shared__ __align__(16) bf16_t sX[256 * 72];     // X^T: [w][ci], stride 72
  __shared__ float sB[192];

  const int b = blockIdx.x;
  const int n = b >> 8, hr = b & 255;
  const int tid = threadIdx.x, lane = tid & 31, wave = tid >> 5;
  const int l = lane & 15, hb = lane >> 4;

  const int rowBase = ((n << 6) << 16) + (hr << 8);  // h[n][c][hr][w]
  if (tid < 64) __builtin_prefetch(&hin[rowBase + (tid << 16)], 0, 0);

  const float* wsrc[3] = {wq, wk, wv};
  const float* bsrc[3] = {bq, bk, bv};
#pragma unroll
  for (int m = 0; m < 3; ++m)
    for (int idx = tid; idx < 4096; idx += 256)
      sW[m * 4608 + (idx >> 6) * 72 + (idx & 63)] = (bf16_t)wsrc[m][idx];
  if (tid < 192) sB[tid] = bsrc[tid >> 6][tid & 63];

  for (int idx = tid; idx < 16384; idx += 256) {
    int c = idx >> 8, ww = idx & 255;                // coalesced over w
    sX[ww * 72 + c] = (bf16_t)hin[rowBase + (c << 16) + ww];
  }
  __syncthreads();

  for (int t = wave; t < 192; t += 8) {              // 3 mats * 4 co * 16 w tiles
    const int m = t >> 6, r = t & 63;
    const int cot = r >> 4, wt = r & 15;
    v8f acc = {};
    acc = wmma_bf16(frag_a(sW + m * 4608, cot << 4, 72, 0),
                    frag_b(sX, wt << 4, 72, 0), acc);
    acc = wmma_bf16(frag_a(sW + m * 4608, cot << 4, 72, 32),
                    frag_b(sX, wt << 4, 72, 32), acc);
    const int cob = (cot << 4) + (hb << 3);
    const int wcol = (wt << 4) + l;
    const float scl = (m == 0) ? 0.0625f : 1.0f;     // fold 1/sqrt(W) into q
    v8bf o;
#pragma unroll
    for (int e = 0; e < 8; ++e)
      o[e] = (bf16_t)((acc[e] + sB[(m << 6) + cob + e]) * scl);
    bf16_t* dst = (m == 0) ? qR : (m == 1) ? kR : vR;
    *(v8bf*)(dst + ((((n << 8) + wcol) << 8) + hr) * 64 + cob) = o;  // [n][w][h][c]
  }
}

// ---------------------------------------------------------------------------
// Kernel 2: row attention for one (n, w) slice: seq = H = 256, dim = C = 64.
// K tile staged async HBM->LDS; V transpose-staged (overlaps the async copy).
// S^T tiles -> per-lane softmax -> P repacked per-lane into A fragments.
// out_r written fp32 to oR in (N, W, H, C).
// ---------------------------------------------------------------------------
__global__ __launch_bounds__(256) void row_attn_kernel(
    const bf16_t* __restrict__ qR, const bf16_t* __restrict__ kR,
    const bf16_t* __restrict__ vR, float* __restrict__ oR) {
  __shared__ __align__(16) char smem_raw[65536];
  bf16_t* sK  = (bf16_t*)smem_raw;                   // [g=256][c=64]
  bf16_t* sVt = (bf16_t*)(smem_raw + 32768);         // [c=64][g=256]

  const int b = blockIdx.x;
  const int n = b >> 8, w = b & 255;
  const int tid = threadIdx.x, lane = tid & 31, wave = tid >> 5;
  const int l = lane & 15, hb = lane >> 4;
  const int sliceBase = (((n << 8) | w) << 14);      // (n*256+w)*256*64

#pragma unroll
  for (int p = 0; p < 8; ++p)                        // K: async 16B chunks
    async_copy16(kR + sliceBase + (p * 256 + tid) * 8, sK + (p * 256 + tid) * 8);
  for (int p = tid; p < 2048; p += 256) {            // V: transpose into LDS
    int g = p >> 3, c8 = (p & 7) << 3;
    v8bf ch = *(const v8bf*)(vR + sliceBase + (g << 6) + c8);
#pragma unroll
    for (int i = 0; i < 8; ++i) sVt[((c8 + i) << 8) + g] = ch[i];
  }
  wait_async0();
  __syncthreads();

#pragma unroll
  for (int ii = 0; ii < 2; ++ii) {
    const int it = wave + (ii << 3);                 // h-tile of 16 rows
    v16bf bq0 = frag_b(qR + sliceBase, it << 4, 64, 0);   // Q^T direct from HBM
    v16bf bq1 = frag_b(qR + sliceBase, it << 4, 64, 32);

    v8f s[16];
    float mx = -3.0e38f;
#pragma unroll
    for (int j = 0; j < 16; ++j) {                   // S^T[g,h] = K @ Q^T
      v8f acc = {};
      acc = wmma_bf16(frag_a(sK, j << 4, 64, 0),  bq0, acc);
      acc = wmma_bf16(frag_a(sK, j << 4, 64, 32), bq1, acc);
      s[j] = acc;
#pragma unroll
      for (int e = 0; e < 8; ++e) mx = fmaxf(mx, acc[e]);
    }
    mx = fmaxf(mx, __shfl_xor(mx, 16));              // partner lane: other g half
    float lsum = 0.f;
#pragma unroll
    for (int j = 0; j < 16; ++j)
#pragma unroll
      for (int e = 0; e < 8; ++e) {
        float t = __expf(s[j][e] - mx);
        s[j][e] = t;
        lsum += t;
      }
    lsum += __shfl_xor(lsum, 16);
    const float inv = 1.0f / lsum;

    v8f o0 = {}, o1 = {}, o2 = {}, o3 = {};
#pragma unroll
    for (int jj = 0; jj < 8; ++jj) {                 // O = P @ V, K-block = 32
      v16bf ap;                                      // per-lane repack D->A
#pragma unroll
      for (int e = 0; e < 8; ++e) {
        ap[e]     = (bf16_t)(s[2 * jj][e] * inv);
        ap[e + 8] = (bf16_t)(s[2 * jj + 1][e] * inv);
      }
      const int g0 = jj << 5;
      o0 = wmma_bf16(ap, frag_b(sVt,  0, 256, g0), o0);
      o1 = wmma_bf16(ap, frag_b(sVt, 16, 256, g0), o1);
      o2 = wmma_bf16(ap, frag_b(sVt, 32, 256, g0), o2);
      o3 = wmma_bf16(ap, frag_b(sVt, 48, 256, g0), o3);
    }
    float* od = oR + sliceBase;                      // same (N,W,H,C) indexing
#pragma unroll
    for (int r = 0; r < 8; ++r) {
      const int hrow = (it << 4) + (hb << 3) + r;
      od[(hrow << 6) +  0 + l] = o0[r];
      od[(hrow << 6) + 16 + l] = o1[r];
      od[(hrow << 6) + 32 + l] = o2[r];
      od[(hrow << 6) + 48 + l] = o3[r];
    }
  }
}

// ---------------------------------------------------------------------------
// Kernel 3: column attention for one (n, h) slice (seq = W), then
// out = out_c + out_r + h  written w-contiguous via an LDS transpose.
// ---------------------------------------------------------------------------
__global__ __launch_bounds__(256) void col_attn_kernel(
    const bf16_t* __restrict__ qR, const bf16_t* __restrict__ kR,
    const bf16_t* __restrict__ vR, const float* __restrict__ oR,
    const float* __restrict__ hin, float* __restrict__ out) {
  __shared__ __align__(16) char smem_raw[65536];
  bf16_t* sK  = (bf16_t*)smem_raw;                   // [u=256][c=64]
  bf16_t* sVt = (bf16_t*)(smem_raw + 32768);         // [c=64][u=256]
  float*  sO  = (float*)smem_raw;                    // reused: [w=256][c=64] f32

  const int b = blockIdx.x;
  const int n = b >> 8, hr = b & 255;
  const int tid = threadIdx.x, lane = tid & 31, wave = tid >> 5;
  const int l = lane & 15, hb = lane >> 4;
  const int tileBase = (n << 22) + (hr << 6);        // (N,W,H,C): + w*16384 + c

#pragma unroll
  for (int p = 0; p < 8; ++p) {                      // K: async 16B chunks
    int idx = p * 256 + tid;
    int wu = idx >> 3, c8 = (idx & 7) << 3;
    async_copy16(kR + tileBase + (wu << 14) + c8, sK + (wu << 6) + c8);
  }
  for (int p = tid; p < 2048; p += 256) {            // V: transpose into LDS
    int wu = p >> 3, c8 = (p & 7) << 3;
    v8bf ch = *(const v8bf*)(vR + tileBase + (wu << 14) + c8);
#pragma unroll
    for (int i = 0; i < 8; ++i) sVt[((c8 + i) << 8) + wu] = ch[i];
  }
  wait_async0();
  __syncthreads();

  v8f oacc[2][4] = {};
#pragma unroll
  for (int ii = 0; ii < 2; ++ii) {
    const int it = wave + (ii << 3);                 // w-tile of 16
    v16bf bq0 = frag_b(qR + tileBase, it << 4, 16384, 0);
    v16bf bq1 = frag_b(qR + tileBase, it << 4, 16384, 32);

    v8f s[16];
    float mx = -3.0e38f;
#pragma unroll
    for (int j = 0; j < 16; ++j) {
      v8f acc = {};
      acc = wmma_bf16(frag_a(sK, j << 4, 64, 0),  bq0, acc);
      acc = wmma_bf16(frag_a(sK, j << 4, 64, 32), bq1, acc);
      s[j] = acc;
#pragma unroll
      for (int e = 0; e < 8; ++e) mx = fmaxf(mx, acc[e]);
    }
    mx = fmaxf(mx, __shfl_xor(mx, 16));
    float lsum = 0.f;
#pragma unroll
    for (int j = 0; j < 16; ++j)
#pragma unroll
      for (int e = 0; e < 8; ++e) {
        float t = __expf(s[j][e] - mx);
        s[j][e] = t;
        lsum += t;
      }
    lsum += __shfl_xor(lsum, 16);
    const float inv = 1.0f / lsum;

#pragma unroll
    for (int jj = 0; jj < 8; ++jj) {
      v16bf ap;
#pragma unroll
      for (int e = 0; e < 8; ++e) {
        ap[e]     = (bf16_t)(s[2 * jj][e] * inv);
        ap[e + 8] = (bf16_t)(s[2 * jj + 1][e] * inv);
      }
      const int g0 = jj << 5;
#pragma unroll
      for (int ct = 0; ct < 4; ++ct)
        oacc[ii][ct] = wmma_bf16(ap, frag_b(sVt, ct << 4, 256, g0), oacc[ii][ct]);
    }
  }
  __syncthreads();                                   // done reading sK/sVt

#pragma unroll
  for (int ii = 0; ii < 2; ++ii) {
    const int it = wave + (ii << 3);
#pragma unroll
    for (int ct = 0; ct < 4; ++ct)
#pragma unroll
      for (int r = 0; r < 8; ++r) {
        const int wrow = (it << 4) + (hb << 3) + r;
        sO[(wrow << 6) + (ct << 4) + l] = oacc[ii][ct][r];
      }
  }
  __syncthreads();

  for (int p = tid; p < 4096; p += 256) {            // += out_r (coalesced f4)
    int wq_ = p >> 4, c4 = (p & 15) << 2;
    const float4 rv = *(const float4*)(oR + tileBase + (wq_ << 14) + c4);
    float* d = sO + (wq_ << 6) + c4;
    d[0] += rv.x; d[1] += rv.y; d[2] += rv.z; d[3] += rv.w;
  }
  __syncthreads();

  for (int p = tid; p < 4096; p += 256) {            // final: + h, w-contiguous
    int c = p >> 6, w0 = (p & 63) << 2;
    int gi = (((n << 6) + c) << 16) + (hr << 8) + w0;
    float4 hv = *(const float4*)(hin + gi);
    float4 ov;
    ov.x = sO[((w0 + 0) << 6) + c] + hv.x;
    ov.y = sO[((w0 + 1) << 6) + c] + hv.y;
    ov.z = sO[((w0 + 2) << 6) + c] + hv.z;
    ov.w = sO[((w0 + 3) << 6) + c] + hv.w;
    *(float4*)(out + gi) = ov;
  }
}

extern "C" void kernel_launch(void* const* d_in, const int* in_sizes, int n_in,
                              void* d_out, int out_size, void* d_ws, size_t ws_size,
                              hipStream_t stream) {
  (void)in_sizes; (void)n_in; (void)out_size; (void)ws_size;
  const float* hin = (const float*)d_in[0];
  const float* wq  = (const float*)d_in[1];
  const float* bq  = (const float*)d_in[2];
  const float* wk  = (const float*)d_in[3];
  const float* bk  = (const float*)d_in[4];
  const float* wv  = (const float*)d_in[5];
  const float* bv  = (const float*)d_in[6];
  float* out = (float*)d_out;

  // Workspace: q/k/v bf16 (N,W,H,C) = 3 * 128 MiB, out_r f32 = 256 MiB.
  const size_t S = (size_t)16 * 256 * 256 * 64;
  bf16_t* qR = (bf16_t*)d_ws;
  bf16_t* kR = qR + S;
  bf16_t* vR = kR + S;
  float*  oR = (float*)(vR + S);

  proj_kernel<<<4096, 256, 0, stream>>>(hin, wq, bq, wk, bk, wv, bv, qR, kR, vR);
  row_attn_kernel<<<4096, 256, 0, stream>>>(qR, kR, vR, oR);
  col_attn_kernel<<<4096, 256, 0, stream>>>(qR, kR, vR, oR, hin, out);
}
